// EGNNLayer_13950053777586
// MI455X (gfx1250) — compile-verified, matrix operand order checked by reading, not run on hardware
//
#include <hip/hip_runtime.h>

#define HH 128
#define NN 512
#define BB 2
#define PADF 132   // padded f32 LDS row stride (banks)
#define SMP  136   // padded f16 LDS row stride (16B aligned, bank-staggered)

typedef __attribute__((ext_vector_type(16))) _Float16 v16h;
typedef __attribute__((ext_vector_type(8)))  _Float16 v8h;
typedef __attribute__((ext_vector_type(8)))  float    v8f;

// fast silu: v * rcp(1+exp(-v))  -> v_exp_f32 + v_rcp_f32, no IEEE div expansion
__device__ __forceinline__ float silu_f(float v) {
  return v * __builtin_amdgcn_rcpf(1.0f + __expf(-v));
}
// fast tanh: (e-1)/(e+1) with e = exp(2x)
__device__ __forceinline__ float tanh_f(float v) {
  float e = __expf(2.0f * v);
  return (e - 1.0f) * __builtin_amdgcn_rcpf(e + 1.0f);
}

// ---------------- weight pre-swizzle into WMMA B-fragment layout (f16) ----------------
// Fragment-major: tile (nt,ks) is 512 halves; within tile, lane l holds 16 contiguous halves.
__global__ void k_swz(const float* __restrict__ eW2, const float* __restrict__ cW1,
                      _Float16* __restrict__ eW2s, _Float16* __restrict__ cW1s) {
  int t = blockIdx.x * blockDim.x + threadIdx.x;   // 0..16383
  int tile = t >> 9, rem = t & 511;
  int l = rem >> 4, hh = rem & 15;
  int nt = tile >> 2, ks = tile & 3;
  int n = nt * 16 + (l & 15);
  int klo = (l < 16) ? 0 : 8;
  int k = ks * 32 + ((hh < 8) ? (klo + hh) : (16 + klo + (hh - 8)));
  eW2s[t] = (_Float16)eW2[k * HH + n];
  cW1s[t] = (_Float16)cW1[k * HH + n];
}

// ---------------- per-node precompute: A = h@eW1[0:128] + eb1 ; Bm = h@eW1[128:256] ----------------
__global__ void k_ab(const float* __restrict__ h, const float* __restrict__ eW1,
                     const float* __restrict__ eb1,
                     float* __restrict__ gA, float* __restrict__ gB) {
  __shared__ float sH[HH];
  int node = blockIdx.x, c = threadIdx.x;
  sH[c] = h[node * HH + c];
  __syncthreads();
  float a = eb1[c], bm = 0.0f;
  #pragma unroll 4
  for (int k = 0; k < HH; ++k) {
    float hv = sH[k];
    a  += hv * eW1[k * HH + c];
    bm += hv * eW1[(HH + k) * HH + c];
  }
  gA[node * HH + c] = a;
  gB[node * HH + c] = bm;
}

// ---------------- fused edge kernel: 2x 128x128 GEMMs per edge tile, all WMMA ----------------
__global__ void __launch_bounds__(256) k_edge(
    const float* __restrict__ x,
    const float* __restrict__ eW1,      // row 256 = dist^2 weight
    const float* __restrict__ eb2,
    const float* __restrict__ cb1,
    const float* __restrict__ cW2,
    const float* __restrict__ gA,
    const float* __restrict__ gB,
    const _Float16* __restrict__ eW2s,
    const _Float16* __restrict__ cW1s,
    float* __restrict__ gMagg,
    float* __restrict__ gCoord)
{
  __shared__ float    sA[16 * PADF];
  __shared__ float    sBm[16 * PADF];
  __shared__ _Float16 sM[128 * SMP];      // m_ij staging (f16) for GEMM2 A-fragments
  __shared__ float    sMagg[16 * HH];
  __shared__ float    sXi[48], sXj[48];
  __shared__ float    sW1d[HH], sEb2[HH], sCb1[HH], sCw2[HH];
  __shared__ float    sCoord[48];

  const int tid   = threadIdx.x;
  const int wave  = tid >> 5;
  const int lane  = tid & 31;
  const int lan15 = lane & 15;
  const int klo   = (lane < 16) ? 0 : 8;

  int bi = blockIdx.x;
  const int JG = 4, NT16 = NN / 16;       // 4 j-groups, 32 i-tiles
  int jg = bi % JG; bi /= JG;
  int it = bi % NT16; bi /= NT16;
  int b  = bi;
  int i0 = it * 16;
  int nodeBase = b * NN;

  for (int idx = tid; idx < 16 * HH; idx += 256) {
    int ii = idx >> 7, c = idx & 127;
    sA[ii * PADF + c] = gA[(nodeBase + i0 + ii) * HH + c];
    sMagg[idx] = 0.0f;
  }
  if (tid < 48) { sXi[tid] = x[(nodeBase + i0) * 3 + tid]; sCoord[tid] = 0.0f; }
  if (tid < HH) {
    sW1d[tid] = eW1[256 * HH + tid];
    sEb2[tid] = eb2[tid];
    sCb1[tid] = cb1[tid];
    sCw2[tid] = cW2[tid];
  }
  __syncthreads();

  for (int q = 0; q < 8; ++q) {
    int j0 = (jg * 8 + q) * 16;
    for (int idx = tid; idx < 16 * HH; idx += 256) {
      int jj = idx >> 7, c = idx & 127;
      sBm[jj * PADF + c] = gB[(nodeBase + j0 + jj) * HH + c];
    }
    if (tid < 48) sXj[tid] = x[(nodeBase + j0) * 3 + tid];
    __syncthreads();

    for (int h2 = 0; h2 < 2; ++h2) {
      __builtin_amdgcn_wave_barrier();
      int ii = h2 * 8 + wave;             // this wave's i within the tile
      // lane geometry: this lane's j within the tile is lan15
      float xd0 = sXi[ii*3+0] - sXj[lan15*3+0];
      float xd1 = sXi[ii*3+1] - sXj[lan15*3+1];
      float xd2 = sXi[ii*3+2] - sXj[lan15*3+2];
      float d2  = xd0*xd0 + xd1*xd1 + xd2*xd2;

      // ---- GEMM1: m = silu( silu(pre1) @ eW2 + eb2 ) ----
      v8f acc[8] = {};
      const float* pA = &sA[ii * PADF];
      const float* pB = &sBm[lan15 * PADF];
      #pragma unroll 1                    // keep ks rolled: bounds register pressure
      for (int ks = 0; ks < 4; ++ks) {
        v16h afrag;
        int c0 = ks * 32 + klo;
        int c1 = ks * 32 + 16 + klo;
        #pragma unroll
        for (int hh = 0; hh < 8; ++hh) {
          float p0 = pA[c0+hh] + pB[c0+hh] + d2 * sW1d[c0+hh];
          float p1 = pA[c1+hh] + pB[c1+hh] + d2 * sW1d[c1+hh];
          afrag[hh]   = (_Float16)silu_f(p0);
          afrag[8+hh] = (_Float16)silu_f(p1);
        }
        const v16h* bp = reinterpret_cast<const v16h*>(eW2s + (ks << 9) + (lane << 4));
        #pragma unroll
        for (int nt = 0; nt < 8; ++nt) {
          v16h bfrag = bp[nt * 128];      // tile stride = 2048 halves = 128 v16h
          acc[nt] = __builtin_amdgcn_wmma_f32_16x16x32_f16(false, afrag, false, bfrag,
                                                           (short)0, acc[nt], false, false);
        }
      }
      // epilogue: +eb2, silu, stage f16 for GEMM2, accumulate m_agg (col sums) in LDS
      #pragma unroll
      for (int nt = 0; nt < 8; ++nt) {
        int n = nt * 16 + lan15;
        float bv = sEb2[n];
        float ssum = 0.0f;
        #pragma unroll
        for (int r = 0; r < 8; ++r) {
          int lrow = wave * 16 + ((lane < 16) ? r : (8 + r));
          float mv = silu_f(acc[nt][r] + bv);
          ssum += mv;
          sM[lrow * SMP + n] = (_Float16)mv;
        }
        atomicAdd(&sMagg[ii * HH + n], ssum);
      }
      __builtin_amdgcn_wave_barrier();    // order wave-local sM stores before loads

      // ---- GEMM2: t = silu( m @ cW1 + cb1 ) ----
      v8f acc2[8] = {};
      int arow = wave * 16 + lan15;
      #pragma unroll 1                    // keep ks rolled here too
      for (int ks = 0; ks < 4; ++ks) {
        v8h lo = *reinterpret_cast<const v8h*>(&sM[arow * SMP + ks * 32 + klo]);
        v8h hi = *reinterpret_cast<const v8h*>(&sM[arow * SMP + ks * 32 + 16 + klo]);
        v16h a2;
        #pragma unroll
        for (int hh = 0; hh < 8; ++hh) { a2[hh] = lo[hh]; a2[8+hh] = hi[hh]; }
        const v16h* bp = reinterpret_cast<const v16h*>(cW1s + (ks << 9) + (lane << 4));
        #pragma unroll
        for (int nt = 0; nt < 8; ++nt) {
          v16h bfrag = bp[nt * 128];
          acc2[nt] = __builtin_amdgcn_wmma_f32_16x16x32_f16(false, a2, false, bfrag,
                                                            (short)0, acc2[nt], false, false);
        }
      }
      // coord weight: cw_j = tanh( dot(t_row, cW2) ), then coord accumulation
      float prt[8] = {};
      #pragma unroll
      for (int nt = 0; nt < 8; ++nt) {
        int n = nt * 16 + lan15;
        float cb = sCb1[n];
        float w  = sCw2[n];
        #pragma unroll
        for (int r = 0; r < 8; ++r)
          prt[r] += silu_f(acc2[nt][r] + cb) * w;
      }
      #pragma unroll
      for (int mask = 1; mask <= 8; mask <<= 1) {
        #pragma unroll
        for (int r = 0; r < 8; ++r)
          prt[r] += __shfl_xor(prt[r], mask, 32);
      }
      if (lan15 == 0) {                    // lanes 0 and 16: rows jj=0..7 / 8..15
        int jbase = (lane < 16) ? 0 : 8;
        float cu0 = 0.f, cu1 = 0.f, cu2 = 0.f;
        #pragma unroll
        for (int r = 0; r < 8; ++r) {
          float cw = tanh_f(prt[r]);
          int jj2 = jbase + r;
          cu0 += (sXi[ii*3+0] - sXj[jj2*3+0]) * cw;
          cu1 += (sXi[ii*3+1] - sXj[jj2*3+1]) * cw;
          cu2 += (sXi[ii*3+2] - sXj[jj2*3+2]) * cw;
        }
        atomicAdd(&sCoord[ii*3+0], cu0);
        atomicAdd(&sCoord[ii*3+1], cu1);
        atomicAdd(&sCoord[ii*3+2], cu2);
      }
    }
    __syncthreads();
  }

  // flush partials (4 j-groups per i-tile -> global atomics)
  for (int idx = tid; idx < 16 * HH; idx += 256) {
    int ii = idx >> 7, c = idx & 127;
    atomicAdd(&gMagg[(nodeBase + i0 + ii) * HH + c], sMagg[idx]);
  }
  if (tid < 48) atomicAdd(&gCoord[(nodeBase + i0) * 3 + tid], sCoord[tid]);
}

// ---------------- node MLP + residual + x update ----------------
__global__ void k_node(const float* __restrict__ h, const float* __restrict__ x,
                       const float* __restrict__ nW1, const float* __restrict__ nb1,
                       const float* __restrict__ nW2, const float* __restrict__ nb2,
                       const float* __restrict__ gMagg, const float* __restrict__ gCoord,
                       float* __restrict__ out) {
  __shared__ float sIn[2 * HH];
  __shared__ float sT[HH];
  int node = blockIdx.x;
  int c = threadIdx.x;
  float hv = h[node * HH + c];
  sIn[c] = hv;
  sIn[HH + c] = gMagg[node * HH + c];
  __syncthreads();
  float a = nb1[c];
  #pragma unroll 4
  for (int k = 0; k < 2 * HH; ++k) a += sIn[k] * nW1[k * HH + c];
  sT[c] = silu_f(a);
  __syncthreads();
  float a2 = nb2[c];
  #pragma unroll 4
  for (int k = 0; k < HH; ++k) a2 += sT[k] * nW2[k * HH + c];
  out[node * HH + c] = hv + a2;
  if (c < 3) {
    out[BB * NN * HH + node * 3 + c] =
        x[node * 3 + c] + gCoord[node * 3 + c] * (1.0f / 511.0f);
  }
}

extern "C" void kernel_launch(void* const* d_in, const int* in_sizes, int n_in,
                              void* d_out, int out_size, void* d_ws, size_t ws_size,
                              hipStream_t stream) {
  const float* h   = (const float*)d_in[0];
  const float* x   = (const float*)d_in[1];
  const float* eW1 = (const float*)d_in[2];
  const float* eb1 = (const float*)d_in[3];
  const float* eW2 = (const float*)d_in[4];
  const float* eb2 = (const float*)d_in[5];
  const float* nW1 = (const float*)d_in[6];
  const float* nb1 = (const float*)d_in[7];
  const float* nW2 = (const float*)d_in[8];
  const float* nb2 = (const float*)d_in[9];
  const float* cW1 = (const float*)d_in[10];
  const float* cb1 = (const float*)d_in[11];
  const float* cW2 = (const float*)d_in[12];
  float* out = (float*)d_out;

  char* ws = (char*)d_ws;
  float*    gA     = (float*)(ws);                 // 1024*128 f32 = 512KB
  float*    gB     = (float*)(ws + 524288);        // 512KB
  float*    gMagg  = (float*)(ws + 1048576);       // 512KB
  float*    gCoord = (float*)(ws + 1572864);       // 12KB
  _Float16* eW2s   = (_Float16*)(ws + 1585152);    // 32KB
  _Float16* cW1s   = (_Float16*)(ws + 1617920);    // 32KB

  hipMemsetAsync(gMagg, 0, 524288 + 12288, stream);      // zero accumulators
  k_swz<<<64, 256, 0, stream>>>(eW2, cW1, eW2s, cW1s);
  k_ab<<<BB * NN, HH, 0, stream>>>(h, eW1, eb1, gA, gB);
  k_edge<<<BB * (NN / 16) * 4, 256, 0, stream>>>(x, eW1, eb2, cb1, cW2,
                                                 gA, gB, eW2s, cW1s, gMagg, gCoord);
  k_node<<<BB * NN, HH, 0, stream>>>(h, x, nW1, nb1, nW2, nb2, gMagg, gCoord, out);
}
